// model2_14877766713699
// MI455X (gfx1250) — matrix operand order repaired and checked by reading, not compiled, not run on hardware
//
#include <hip/hip_runtime.h>
#include <hip/hip_bf16.h>

typedef __attribute__((ext_vector_type(16))) _Float16 v16h;
typedef __attribute__((ext_vector_type(8)))  float    v8f;

#define HID   8
#define INSZ  11
#define SEQ   512
#define BATCH 16384

__device__ __forceinline__ float fast_rcp(float x) {
#if __has_builtin(__builtin_amdgcn_rcpf)
    return __builtin_amdgcn_rcpf(x);      // raw v_rcp_f32, no div-fixup
#else
    return 1.0f / x;
#endif
}

__device__ __forceinline__ float fast_tanh(float x) {
#if __has_builtin(__builtin_amdgcn_tanhf)
    return __builtin_amdgcn_tanhf(x);     // v_tanh_f32 (gfx1250 TRANS op)
#elif __has_builtin(__builtin_amdgcn_tanh_f32)
    return __builtin_amdgcn_tanh_f32(x);
#else
    float e = __expf(2.0f * x);
    return 1.0f - 2.0f * fast_rcp(e + 1.0f);
#endif
}

__device__ __forceinline__ float fast_sigmoid(float x) {
    // sigmoid(x) = 0.5 + 0.5*tanh(x/2): mul + tanh + fma, no division
    return __fmaf_rn(0.5f, fast_tanh(0.5f * x), 0.5f);
}

__device__ __forceinline__ float shflx16(float v) {
    return __shfl_xor(v, 16, 32);
}

// One wave (32 lanes) owns a 16-wide batch tile (N dimension of every WMMA).
// State h1/c1/h2/c2 lives in lanes 0-15 (lane = batch column, reg = hidden idx).
__global__ __launch_bounds__(256)
void lstm2_fused_kernel(const float* __restrict__ x,
                        const float* __restrict__ W_ih1, const float* __restrict__ W_hh1,
                        const float* __restrict__ b_ih1, const float* __restrict__ b_hh1,
                        const float* __restrict__ W_ih2, const float* __restrict__ W_hh2,
                        const float* __restrict__ b_ih2, const float* __restrict__ b_hh2,
                        const float* __restrict__ fc1_w, const float* __restrict__ fc1_b,
                        const float* __restrict__ fc2_w, const float* __restrict__ fc2_b,
                        float* __restrict__ out)
{
    const int tid  = blockIdx.x * blockDim.x + threadIdx.x;
    const int wave = tid >> 5;
    const int lane = threadIdx.x & 31;
    const int col  = lane & 15;   // N = batch column within tile
    const int hi   = lane >> 4;   // lane half
    const int bidx = wave * 16 + col;

    // ---- Build constant A operands (weights ++ bias column, padded to K=32) ----
    // A 16x32, contiguous k order used here (B1/B2 match): lane L holds row M=L&15; half idx j -> K:
    //   K = ((L>>4)<<4) | j
    v16h a1lo, a1hi, a2lo, a2hi;
    const int M = col;
    #pragma unroll
    for (int j = 0; j < 16; ++j) {
        const int K = (hi << 4) | j;
        // Layer 1: Wcat1[32 x 32] = [W_ih1 | W_hh1 | b_ih1+b_hh1 | 0]
        float w1l = 0.f, w1h = 0.f;
        if (K < INSZ)                { w1l = W_ih1[M*INSZ + K];          w1h = W_ih1[(M+16)*INSZ + K]; }
        else if (K < INSZ + HID)     { w1l = W_hh1[M*HID + (K-INSZ)];    w1h = W_hh1[(M+16)*HID + (K-INSZ)]; }
        else if (K == INSZ + HID)    { w1l = b_ih1[M] + b_hh1[M];        w1h = b_ih1[M+16] + b_hh1[M+16]; }
        a1lo[j] = (_Float16)w1l;  a1hi[j] = (_Float16)w1h;
        // Layer 2: Wcat2[32 x 32] = [W_ih2 | W_hh2 | b_ih2+b_hh2 | 0]
        float w2l = 0.f, w2h = 0.f;
        if (K < HID)                 { w2l = W_ih2[M*HID + K];           w2h = W_ih2[(M+16)*HID + K]; }
        else if (K < 2*HID)          { w2l = W_hh2[M*HID + (K-HID)];     w2h = W_hh2[(M+16)*HID + (K-HID)]; }
        else if (K == 2*HID)         { w2l = b_ih2[M] + b_hh2[M];        w2h = b_ih2[M+16] + b_hh2[M+16]; }
        a2lo[j] = (_Float16)w2l;  a2hi[j] = (_Float16)w2h;
    }

    float h1[HID], c1[HID], h2[HID], c2[HID];
    #pragma unroll
    for (int r = 0; r < HID; ++r) { h1[r] = 0.f; c1[r] = 0.f; h2[r] = 0.f; c2[r] = 0.f; }

    const v8f vzero = {};
    // dhi tiles: lo lanes hold g (want tanh(v)); hi lanes hold o (want sigmoid(v)
    // = 0.5 + 0.5*tanh(0.5*v)).  One tanh per value: scale arg per half.
    const float hsc = hi ? 0.5f : 1.0f;

    auto step = [&](int t) {
        const float* xt = x + ((size_t)t * BATCH + bidx) * INSZ;
        // broadcast h1[5..7] to the high half-lanes (they carry K=16..18 of B1)
        float p5 = shflx16(h1[5]);
        float p6 = shflx16(h1[6]);
        float p7 = shflx16(h1[7]);

        // ---- B1 = [x(0..10); h1(0..7); 1; pad]  (K x N, contiguous k order, same as A:
        //       lanes 0-15: j -> K=j ; lanes 16-31: j -> K=16+j)
        v16h B1 = {};
        if (hi == 0) {
            B1[0]  = (_Float16)xt[0];  B1[1]  = (_Float16)xt[1];
            B1[2]  = (_Float16)xt[2];  B1[3]  = (_Float16)xt[3];
            B1[4]  = (_Float16)xt[4];  B1[5]  = (_Float16)xt[5];
            B1[6]  = (_Float16)xt[6];  B1[7]  = (_Float16)xt[7];
            B1[8]  = (_Float16)xt[8];  B1[9]  = (_Float16)xt[9];
            B1[10] = (_Float16)xt[10];
            B1[11] = (_Float16)h1[0];  B1[12] = (_Float16)h1[1];
            B1[13] = (_Float16)h1[2];  B1[14] = (_Float16)h1[3];
            B1[15] = (_Float16)h1[4];
            // prefetch next timestep's row for this batch column
            __builtin_prefetch(xt + (size_t)BATCH * INSZ, 0, 3);
        } else {
            B1[0] = (_Float16)p5;      // K=16 -> h1[5]
            B1[1] = (_Float16)p6;      // K=17 -> h1[6]
            B1[2] = (_Float16)p7;      // K=18 -> h1[7]
            B1[3] = (_Float16)1.0f;    // K=19 -> bias row
        }

        // ---- layer-1 gates: G1[32x16] = Wcat1 @ B1 (two 16x16 tiles) ----
        v8f dlo = __builtin_amdgcn_wmma_f32_16x16x32_f16(false, a1lo, false, B1,
                                                         (short)0, vzero, false, false);
        v8f dhi = __builtin_amdgcn_wmma_f32_16x16x32_f16(false, a1hi, false, B1,
                                                         (short)0, vzero, false, false);
        // D layout: lanes 0-15 hold rows r (i / g), lanes 16-31 rows r+8 (f / o)
        #pragma unroll
        for (int r = 0; r < HID; ++r) {
            float s  = fast_sigmoid(dlo[r]);        // i (lo lanes) / f (hi lanes)
            float fs = shflx16(s);                  // lo lanes: sigmoid(f)
            float tt = fast_tanh(hsc * dhi[r]);     // lo: tanh(g); hi: tanh(o/2)
            float sg = __fmaf_rn(0.5f, tt, 0.5f);   // hi lanes: sigmoid(o)
            float op = shflx16(sg);                 // lo lanes: sigmoid(o)
            float cn = fs * c1[r] + s * tt;
            c1[r] = cn;
            h1[r] = op * fast_tanh(cn);
        }

        // ---- B2 = [h1(0..7); h2(0..7); 1; pad] ----
        v16h B2 = {};
        if (hi == 0) {
            B2[0]  = (_Float16)h1[0];  B2[1]  = (_Float16)h1[1];
            B2[2]  = (_Float16)h1[2];  B2[3]  = (_Float16)h1[3];
            B2[4]  = (_Float16)h1[4];  B2[5]  = (_Float16)h1[5];
            B2[6]  = (_Float16)h1[6];  B2[7]  = (_Float16)h1[7];
            B2[8]  = (_Float16)h2[0];  B2[9]  = (_Float16)h2[1];
            B2[10] = (_Float16)h2[2];  B2[11] = (_Float16)h2[3];
            B2[12] = (_Float16)h2[4];  B2[13] = (_Float16)h2[5];
            B2[14] = (_Float16)h2[6];  B2[15] = (_Float16)h2[7];
        } else {
            B2[0] = (_Float16)1.0f;    // K=16 -> bias row
        }

        v8f elo = __builtin_amdgcn_wmma_f32_16x16x32_f16(false, a2lo, false, B2,
                                                         (short)0, vzero, false, false);
        v8f ehi = __builtin_amdgcn_wmma_f32_16x16x32_f16(false, a2hi, false, B2,
                                                         (short)0, vzero, false, false);
        #pragma unroll
        for (int r = 0; r < HID; ++r) {
            float s  = fast_sigmoid(elo[r]);
            float fs = shflx16(s);
            float tt = fast_tanh(hsc * ehi[r]);
            float sg = __fmaf_rn(0.5f, tt, 0.5f);
            float op = shflx16(sg);
            float cn = fs * c2[r] + s * tt;
            c2[r] = cn;
            h2[r] = op * fast_tanh(cn);
        }
    };

    for (int t = 0; t < SEQ; ++t) step(t);
    step(4);  // reference's extra hard-coded step on input[4]

    // ---- head: out = fc2 @ (fc1 @ h2 + fc1_b) + fc2_b ----
    if (hi == 0) {
        float acc = fc2_b[0];
        #pragma unroll
        for (int j = 0; j < 4; ++j) {
            float y = fc1_b[j];
            #pragma unroll
            for (int k = 0; k < HID; ++k) y += fc1_w[j * HID + k] * h2[k];
            acc += fc2_w[j] * y;
        }
        out[bidx] = acc;
    }
}

extern "C" void kernel_launch(void* const* d_in, const int* in_sizes, int n_in,
                              void* d_out, int out_size, void* d_ws, size_t ws_size,
                              hipStream_t stream) {
    (void)in_sizes; (void)n_in; (void)out_size; (void)d_ws; (void)ws_size;
    const float* x     = (const float*)d_in[0];
    const float* W_ih1 = (const float*)d_in[1];
    const float* W_hh1 = (const float*)d_in[2];
    const float* b_ih1 = (const float*)d_in[3];
    const float* b_hh1 = (const float*)d_in[4];
    const float* W_ih2 = (const float*)d_in[5];
    const float* W_hh2 = (const float*)d_in[6];
    const float* b_ih2 = (const float*)d_in[7];
    const float* b_hh2 = (const float*)d_in[8];
    const float* fc1_w = (const float*)d_in[9];
    const float* fc1_b = (const float*)d_in[10];
    const float* fc2_w = (const float*)d_in[11];
    const float* fc2_b = (const float*)d_in[12];
    float* out = (float*)d_out;

    // 16 batch columns per wave, 8 waves (256 threads) per block
    const int waves  = BATCH / 16;           // 1024
    const int blocks = waves / 8;            // 128
    lstm2_fused_kernel<<<blocks, 256, 0, stream>>>(
        x, W_ih1, W_hh1, b_ih1, b_hh1, W_ih2, W_hh2, b_ih2, b_hh2,
        fc1_w, fc1_b, fc2_w, fc2_b, out);
}
